// RegionFeature_91104846282871
// MI455X (gfx1250) — matrix operand order repaired and checked by reading, not compiled, not run on hardware
//
#include <hip/hip_runtime.h>
#include <hip/hip_bf16.h>
#include <math.h>

typedef __attribute__((ext_vector_type(16))) _Float16 v16h;
typedef __attribute__((ext_vector_type(8)))  _Float16 v8h;
typedef __attribute__((ext_vector_type(8)))  float    v8f;

#define NEGV      (-1e30f)
#define NMS_THR   0.7f
#define MIN_SZ    0.001f
#define KTOT      3507
#define POST      300
#define BBOX_CLMP 4.135166556742356f   // log(1000/16)

__device__ __forceinline__ v8f wmma16(v16h a, v16h b, v8f c) {
  // D = A(16x32 f16) * B(32x16 f16) + C(f32)
  return __builtin_amdgcn_wmma_f32_16x16x32_f16(
      /*neg_a=*/false, a, /*neg_b=*/false, b,
      /*c_mod=*/(short)0, c, /*reuse_a=*/false, /*reuse_b=*/false);
}

// ---------------------------------------------------------------------------
// Weight packing: conv_w (co,ci,3,3) f32 -> per-lane fragment-ready f16.
// Layout: out[((kc*16 + t)*32 + lane)*16 + hh]
//   kc = tap*8 + cb (72 K-chunks of 32), t = co/16 tile, lane = wave lane,
//   hh = half index inside the v16h A-fragment.
// ---------------------------------------------------------------------------
__global__ void cvt_conv_w_kernel(const float* __restrict__ w, _Float16* __restrict__ out, int n) {
  int i = blockIdx.x * 256 + threadIdx.x;
  if (i >= n) return;
  int hh = i & 15;
  int l  = (i >> 4) & 31;
  int t  = (i >> 9) & 15;
  int kc = i >> 13;                    // 0..71
  int tap = kc >> 3, cb = kc & 7;
  int lanec = l >> 4, lanem = l & 15;
  int v = hh >> 1, r = hh & 1;
  int kk = lanec * 8 + (v & 3) * 2 + ((v >> 2) << 4) + r;  // ISA 16-bit A layout
  int ci = cb * 32 + kk;
  int co = t * 16 + lanem;
  out[i] = (_Float16)w[(co * 256 + ci) * 9 + tap];
}

__global__ void cvt_fc_w_kernel(const float* __restrict__ w, _Float16* __restrict__ out, long n) {
  long i = (long)blockIdx.x * 256 + threadIdx.x;
  if (i < n) out[i] = (_Float16)w[i];
}

// NCHW f32 -> spatially padded (+1 ring) NHWC f16: fp[((b*(H+2)+yy)*(W+2)+xx)*256+c]
__global__ void cvt_featpad_kernel(const float* __restrict__ feat, _Float16* __restrict__ fp,
                                   int H, int W, long total) {
  long g = (long)blockIdx.x * 256 + threadIdx.x;
  if (g >= total) return;
  int c = (int)(g & 255);
  long pix = g >> 8;
  int Wp = W + 2, Hp = H + 2;
  int xx = (int)(pix % Wp);
  long r2 = pix / Wp;
  int yy = (int)(r2 % Hp);
  int b  = (int)(r2 / Hp);
  float v = 0.f;
  if (yy >= 1 && yy <= H && xx >= 1 && xx <= W)
    v = feat[(((long)b * 256 + c) * H + (yy - 1)) * W + (xx - 1)];
  fp[g] = (_Float16)v;
}

// ---------------------------------------------------------------------------
// 3x3 conv + bias + relu as implicit GEMM (WMMA)
//   M = out channels (64/block), N = pixels (64/block), K = 9*256
//   A fragments: direct v16h loads from packed weights (no LDS).
//   B tile: one b128 global load + one b128 LDS store per thread per chunk;
//           fragments: 2x ds_load_b128 per operand.
// ---------------------------------------------------------------------------
__global__ __launch_bounds__(256) void conv3x3_wmma_kernel(
    const _Float16* __restrict__ featp,  // padded NHWC f16
    const v16h*     __restrict__ wpack,  // fragment-packed weights
    const float*    __restrict__ bias,
    _Float16* __restrict__ tout,
    int H, int W, int HW) {
  __shared__ _Float16 Bsf[64][40];       // [pixel][kk], K-major per pixel
  int tid  = threadIdx.x;
  int b    = blockIdx.z;
  int mblk = blockIdx.y * 64;            // out-channel tile
  int nblk = blockIdx.x * 64;            // pixel tile
  int wv = tid >> 5, lane = tid & 31;
  int lanec = lane >> 4, lanem = lane & 15;
  int mi = wv & 3, nj = wv >> 2;
  int m0 = mi * 16, n0a = nj * 32, n0b = n0a + 16;
  int tA = (mblk >> 4) + mi;             // global co/16 tile for this wave
  // staging assignment for this thread
  int sn = tid & 63, skk0 = (tid >> 6) * 8;
  int sp = nblk + sn;
  int sy = 0, sx = 0;
  if (sp < HW) { sy = sp / W; sx = sp - sy * W; }
  int Wp = W + 2;
  v8f acc0 = {}, acc1 = {};

  #pragma unroll 1
  for (int kc = 0; kc < 72; ++kc) {
    int tap = kc >> 3, cb = kc & 7;
    int dy = tap / 3, dx = tap - (tap / 3) * 3;   // padded offsets 0..2
    v8h vv = {};
    if (sp < HW) {
      long o = ((((long)b * (H + 2) + (sy + dy)) * Wp) + (sx + dx)) * 256 + cb * 32 + skk0;
      vv = *(const v8h*)(featp + o);
      __builtin_prefetch((const void*)(featp + o + 32), 0, 0);  // next cb chunk
    }
    *(v8h*)(&Bsf[sn][skk0]) = vv;
    __syncthreads();

    v16h af = wpack[(kc * 16 + tA) * 32 + lane];
    v16h bf0, bf1;
    {
      const v8h* p0 = (const v8h*)(&Bsf[n0a + lanem][lanec * 8]);
      ((v8h*)&bf0)[0] = p0[0];
      ((v8h*)&bf0)[1] = p0[2];
      const v8h* p1 = (const v8h*)(&Bsf[n0b + lanem][lanec * 8]);
      ((v8h*)&bf1)[0] = p1[0];
      ((v8h*)&bf1)[1] = p1[2];
    }
    acc0 = wmma16(af, bf0, acc0);
    acc1 = wmma16(af, bf1, acc1);
    __syncthreads();
  }

  // epilogue: bias + relu, store f16 t (coalesced along pixels)
  #pragma unroll
  for (int r = 0; r < 8; ++r) {
    int co = mblk + m0 + r + lanec * 8;
    float bv = bias[co];
    int p0 = nblk + n0a + lanem;
    if (p0 < HW) tout[((long)b * 256 + co) * HW + p0] = (_Float16)fmaxf(acc0[r] + bv, 0.f);
    int p1 = nblk + n0b + lanem;
    if (p1 < HW) tout[((long)b * 256 + co) * HW + p1] = (_Float16)fmaxf(acc1[r] + bv, 0.f);
  }
}

// ---------------------------------------------------------------------------
// 1x1 heads: 15 units (3 logits + 12 deltas) per pixel
// ---------------------------------------------------------------------------
__global__ void head1x1_kernel(const _Float16* __restrict__ t,
    const float* __restrict__ cls_w, const float* __restrict__ cls_b,
    const float* __restrict__ bbox_w, const float* __restrict__ bbox_b,
    float* __restrict__ logits, float* __restrict__ deltas, int HW, int total) {
  int g = blockIdx.x * 256 + threadIdx.x;
  if (g >= total) return;
  int pix = g % HW;
  int u   = (g / HW) % 15;
  int b   = g / (HW * 15);
  const float* wr = (u < 3) ? (cls_w + u * 256) : (bbox_w + (u - 3) * 256);
  float s = (u < 3) ? cls_b[u] : bbox_b[u - 3];
  const _Float16* tb = t + (long)b * 256 * HW + pix;
  for (int c = 0; c < 256; ++c) s += (float)tb[(long)c * HW] * wr[c];
  if (u < 3) {
    logits[(long)b * 3 * HW + (long)pix * 3 + u] = s;
  } else {
    int q = u - 3, a = q >> 2, d = q & 3;
    deltas[((long)b * 3 * HW + (long)pix * 3 + a) * 4 + d] = s;
  }
}

// ---------------------------------------------------------------------------
// top-k via iterative argmax (one block per (lvl, b)); mutates logits buffer
// ---------------------------------------------------------------------------
struct TkParams { long loff[4]; int N[4]; int K[4]; int koff[4]; };

__global__ __launch_bounds__(1024) void topk_kernel(float* __restrict__ logits_all,
    TkParams tp, float* __restrict__ tscore, int* __restrict__ tidx) {
  int lvl = blockIdx.x, b = blockIdx.y;
  int N = tp.N[lvl], K = tp.K[lvl], ko = tp.koff[lvl];
  float* sc = logits_all + tp.loff[lvl] + (long)b * N;
  __shared__ float rv[1024]; __shared__ int ri[1024];
  int tid = threadIdx.x;
  for (int it = 0; it < K; ++it) {
    float best = -INFINITY; int bi = 0x7fffffff;
    for (int i = tid; i < N; i += 1024) {
      float v = sc[i];
      if (v > best || (v == best && i < bi)) { best = v; bi = i; }
    }
    rv[tid] = best; ri[tid] = bi;
    __syncthreads();
    for (int s = 512; s > 0; s >>= 1) {
      if (tid < s) {
        if (rv[tid + s] > rv[tid] || (rv[tid + s] == rv[tid] && ri[tid + s] < ri[tid])) {
          rv[tid] = rv[tid + s]; ri[tid] = ri[tid + s];
        }
      }
      __syncthreads();
    }
    if (tid == 0) {
      int j = ri[0];
      tscore[(long)b * KTOT + ko + it] = rv[0];
      tidx[(long)b * KTOT + ko + it]   = j;
      sc[j] = -INFINITY;
    }
    __syncthreads();
  }
}

// ---------------------------------------------------------------------------
// anchor decode + clip + filter
// ---------------------------------------------------------------------------
struct DecParams { long doff[4]; int H[4]; int W[4]; float size[4]; };

__global__ void decode_kernel(const float* __restrict__ tscore, const int* __restrict__ tidx,
    const float* __restrict__ deltas_all, DecParams dp,
    const int* __restrict__ img_h_p, const int* __restrict__ img_w_p,
    float* __restrict__ cscore, float* __restrict__ cbox, float* __restrict__ cbox_off,
    int total) {
  int g = blockIdx.x * 256 + threadIdx.x;
  if (g >= total) return;
  int b = g / KTOT, p = g % KTOT;
  int lvl = p / 1000; if (lvl > 3) lvl = 3;
  int H = dp.H[lvl], W = dp.W[lvl];
  int HW = H * W;
  int img_h = *img_h_p, img_w = *img_w_p;
  float stride = (float)(img_h / H);           // python: float(img_h // H)
  int idx = tidx[g];
  float ts = tscore[g];
  int a = idx % 3; int cell = idx / 3;
  int x = cell % W, y = cell / W;
  const float ratios[3] = {0.5f, 1.f, 2.f};
  float hr = sqrtf(ratios[a]); float wr = 1.f / hr;
  float wsz = wr * dp.size[lvl], hsz = hr * dp.size[lvl];
  float ax1 = rintf(-wsz * 0.5f) + x * stride;
  float ay1 = rintf(-hsz * 0.5f) + y * stride;
  float ax2 = rintf( wsz * 0.5f) + x * stride;
  float ay2 = rintf( hsz * 0.5f) + y * stride;
  float wa = ax2 - ax1, ha = ay2 - ay1;
  float cxa = ax1 + 0.5f * wa, cya = ay1 + 0.5f * ha;
  const float* dl = deltas_all + dp.doff[lvl] + ((long)b * 3 * HW + idx) * 4;
  float dxv = dl[0], dyv = dl[1];
  float dw = fminf(dl[2], BBOX_CLMP);
  float dh = fminf(dl[3], BBOX_CLMP);
  float cx = dxv * wa + cxa, cy = dyv * ha + cya;
  float w = expf(dw) * wa, h = expf(dh) * ha;
  float x1 = fminf(fmaxf(cx - 0.5f * w, 0.f), (float)img_w);
  float y1 = fminf(fmaxf(cy - 0.5f * h, 0.f), (float)img_h);
  float x2 = fminf(fmaxf(cx + 0.5f * w, 0.f), (float)img_w);
  float y2 = fminf(fmaxf(cy + 0.5f * h, 0.f), (float)img_h);
  if ((x2 - x1) < MIN_SZ || (y2 - y1) < MIN_SZ || ts < 0.f) ts = NEGV;
  long o = (long)g * 4;
  cbox[o] = x1; cbox[o + 1] = y1; cbox[o + 2] = x2; cbox[o + 3] = y2;
  float off = lvl * 4096.f;
  cbox_off[o] = x1 + off; cbox_off[o + 1] = y1 + off;
  cbox_off[o + 2] = x2 + off; cbox_off[o + 3] = y2 + off;
  cscore[g] = ts;
}

// ---------------------------------------------------------------------------
// sequential NMS (one block per batch image)
// ---------------------------------------------------------------------------
__global__ __launch_bounds__(1024) void nms_kernel(
    const float* __restrict__ cscore, const float* __restrict__ cbox_off,
    int* __restrict__ oidx, int* __restrict__ oval) {
  int b = blockIdx.x; int tid = threadIdx.x;
  __shared__ float sc[KTOT];
  __shared__ float rv[1024]; __shared__ int ri[1024];
  __shared__ float bj[5];    __shared__ int jjs;
  const float* boxes = cbox_off + (long)b * KTOT * 4;
  for (int i = tid; i < KTOT; i += 1024) sc[i] = cscore[(long)b * KTOT + i];
  __syncthreads();
  for (int it = 0; it < POST; ++it) {
    float best = -INFINITY; int bi = 0x7fffffff;
    for (int i = tid; i < KTOT; i += 1024) {
      float v = sc[i];
      if (v > best || (v == best && i < bi)) { best = v; bi = i; }
    }
    rv[tid] = best; ri[tid] = bi;
    __syncthreads();
    for (int s = 512; s > 0; s >>= 1) {
      if (tid < s) {
        if (rv[tid + s] > rv[tid] || (rv[tid + s] == rv[tid] && ri[tid + s] < ri[tid])) {
          rv[tid] = rv[tid + s]; ri[tid] = ri[tid + s];
        }
      }
      __syncthreads();
    }
    if (tid == 0) {
      int j = ri[0]; jjs = j;
      oidx[b * POST + it] = j;
      oval[b * POST + it] = (sc[j] > NEGV * 0.5f) ? 1 : 0;
      float x1 = boxes[j * 4], y1 = boxes[j * 4 + 1];
      float x2 = boxes[j * 4 + 2], y2 = boxes[j * 4 + 3];
      bj[0] = x1; bj[1] = y1; bj[2] = x2; bj[3] = y2;
      bj[4] = (x2 - x1) * (y2 - y1);
    }
    __syncthreads();
    int j = jjs;
    for (int i = tid; i < KTOT; i += 1024) {
      float x1 = boxes[i * 4], y1 = boxes[i * 4 + 1];
      float x2 = boxes[i * 4 + 2], y2 = boxes[i * 4 + 3];
      float ai = (x2 - x1) * (y2 - y1);
      float xx1 = fmaxf(bj[0], x1), yy1 = fmaxf(bj[1], y1);
      float xx2 = fminf(bj[2], x2), yy2 = fminf(bj[3], y2);
      float inter = fmaxf(xx2 - xx1, 0.f) * fmaxf(yy2 - yy1, 0.f);
      float iou = inter / fmaxf(bj[4] + ai - inter, 1e-9f);
      if (iou > NMS_THR) sc[i] = NEGV;
      if (i == j) sc[i] = NEGV;
    }
    __syncthreads();
  }
}

__global__ void gather_rois_kernel(const float* __restrict__ cbox,
    const int* __restrict__ oidx, const int* __restrict__ oval,
    float* __restrict__ rois) {
  int i = blockIdx.x * 256 + threadIdx.x;
  if (i >= 2 * POST) return;
  int b = i / POST;
  int j = oidx[i];
  int v = oval[i];
  #pragma unroll
  for (int d = 0; d < 4; ++d)
    rois[i * 4 + d] = v ? cbox[((long)b * KTOT + j) * 4 + d] : 0.f;
}

// zero the padding rows of the ROI-feature matrix (rows 600..639)
__global__ void zero_rfpad_kernel(_Float16* __restrict__ rf, long start, long n) {
  long i = (long)blockIdx.x * 256 + threadIdx.x;
  if (i < n) rf[start + i] = (_Float16)0.f;
}

// ---------------------------------------------------------------------------
// ROI align on feat3 (7x7 out, 2x2 sampling); output f16 [roi][c*49+pix]
// ---------------------------------------------------------------------------
__global__ __launch_bounds__(256) void roi_align_kernel(
    const float* __restrict__ feat3, const float* __restrict__ rois,
    const int* __restrict__ img_h_p, _Float16* __restrict__ rf,
    int Hf, int C) {
  int r = blockIdx.x;
  int b = r / POST;
  float scale = (float)Hf / (float)(*img_h_p);
  __shared__ float syl[14]; __shared__ int syi0[14], syi1[14], svy[14];
  __shared__ float sxl[14]; __shared__ int sxi0[14], sxi1[14], svx[14];
  int tid = threadIdx.x;
  if (tid < 28) {
    int i = tid % 14; bool isy = tid < 14;
    float c1 = rois[r * 4 + (isy ? 1 : 0)] * scale - 0.5f;
    float c2 = rois[r * 4 + (isy ? 3 : 2)] * scale - 0.5f;
    float g = ((float)i + 0.5f) / 14.f;
    float v = c1 + g * (c2 - c1);
    int valid = (v >= -1.f && v <= (float)Hf) ? 1 : 0;
    float vc = fminf(fmaxf(v, 0.f), (float)Hf - 1.f);
    int t0 = (int)floorf(vc);
    int t1 = t0 + 1 < Hf - 1 ? t0 + 1 : Hf - 1;
    float l = vc - (float)t0;
    if (isy) { syl[i] = l; syi0[i] = t0; syi1[i] = t1; svy[i] = valid; }
    else     { sxl[i] = l; sxi0[i] = t0; sxi1[i] = t1; svx[i] = valid; }
  }
  __syncthreads();
  for (int work = tid; work < C * 49; work += 256) {
    int c = work / 49;
    int op = work % 49;
    int oy = op / 7, ox = op % 7;
    const float* fm = feat3 + ((long)b * C + c) * Hf * Hf;
    float s = 0.f;
    #pragma unroll
    for (int sy = 0; sy < 2; ++sy)
      #pragma unroll
      for (int sx = 0; sx < 2; ++sx) {
        int iy = oy * 2 + sy, ix = ox * 2 + sx;
        if (svy[iy] && svx[ix]) {
          int y0 = syi0[iy], y1 = syi1[iy], x0 = sxi0[ix], x1 = sxi1[ix];
          float ly = syl[iy], lx = sxl[ix];
          float c00 = fm[y0 * Hf + x0], c01 = fm[y0 * Hf + x1];
          float c10 = fm[y1 * Hf + x0], c11 = fm[y1 * Hf + x1];
          s += c00 * (1.f - ly) * (1.f - lx) + c01 * (1.f - ly) * lx
             + c10 * ly * (1.f - lx)        + c11 * ly * lx;
        }
      }
    rf[(long)r * 12544 + work] = (_Float16)(s * 0.25f);
  }
}

// ---------------------------------------------------------------------------
// final FC: out[m][n] = sum_k A[m][k] * W[n][k] + b[n]   (WMMA, LDS-free:
// both operands are K-contiguous per lane -> aligned b128 global loads)
// ---------------------------------------------------------------------------
__global__ __launch_bounds__(256) void fc_wmma_kernel(
    const _Float16* __restrict__ A,    // [640][K], rows >= M zeroed
    const _Float16* __restrict__ Bw,   // [N][K]
    const float* __restrict__ bias, float* __restrict__ out,
    int M, int N, int K) {
  int tid  = threadIdx.x;
  int mblk = blockIdx.x * 64, nblk = blockIdx.y * 64;
  int wv = tid >> 5, lane = tid & 31;
  int lanec = lane >> 4, lanem = lane & 15;
  int mi = wv & 3, nj = wv >> 2;
  int m0 = mi * 16, n0a = nj * 32, n0b = n0a + 16;
  const _Float16* arow  = A  + (long)(mblk + m0 + lanem) * K + lanec * 16;
  const _Float16* brow0 = Bw + (long)(nblk + n0a + lanem) * K + lanec * 16;
  const _Float16* brow1 = Bw + (long)(nblk + n0b + lanem) * K + lanec * 16;
  v8f acc0 = {}, acc1 = {};
  #pragma unroll 2
  for (int k0 = 0; k0 < K; k0 += 32) {
    v16h af, bf0, bf1;
    ((v8h*)&af)[0]  = *(const v8h*)(arow + k0);
    ((v8h*)&af)[1]  = *(const v8h*)(arow + k0 + 8);
    ((v8h*)&bf0)[0] = *(const v8h*)(brow0 + k0);
    ((v8h*)&bf0)[1] = *(const v8h*)(brow0 + k0 + 8);
    ((v8h*)&bf1)[0] = *(const v8h*)(brow1 + k0);
    ((v8h*)&bf1)[1] = *(const v8h*)(brow1 + k0 + 8);
    acc0 = wmma16(af, bf0, acc0);
    acc1 = wmma16(af, bf1, acc1);
  }
  #pragma unroll
  for (int r = 0; r < 8; ++r) {
    int m = mblk + m0 + r + lanec * 8;
    if (m < M) {
      int nA = nblk + n0a + lanem;
      out[(long)m * N + nA] = acc0[r] + bias[nA];
      int nB = nblk + n0b + lanem;
      out[(long)m * N + nB] = acc1[r] + bias[nB];
    }
  }
}

// ---------------------------------------------------------------------------
// host orchestration
// ---------------------------------------------------------------------------
extern "C" void kernel_launch(void* const* d_in, const int* in_sizes, int n_in,
                              void* d_out, int out_size, void* d_ws, size_t ws_size,
                              hipStream_t stream) {
  const float* feats[4] = {(const float*)d_in[0], (const float*)d_in[1],
                           (const float*)d_in[2], (const float*)d_in[3]};
  const float* conv_w = (const float*)d_in[4];
  const float* conv_b = (const float*)d_in[5];
  const float* cls_w  = (const float*)d_in[6];
  const float* cls_b  = (const float*)d_in[7];
  const float* bbox_w = (const float*)d_in[8];
  const float* bbox_b = (const float*)d_in[9];
  const float* fc_w   = (const float*)d_in[10];
  const float* fc_b   = (const float*)d_in[11];
  const int*   img_h  = (const int*)d_in[12];
  const int*   img_w  = (const int*)d_in[13];

  const int B = 2;
  const int Hs[4] = {104, 52, 26, 13};
  const float SIZES[4] = {32.f, 64.f, 128.f, 256.f};
  int HW[4], N3[4], Kk[4], koff[4];
  int ko = 0;
  for (int l = 0; l < 4; ++l) {
    HW[l] = Hs[l] * Hs[l];
    N3[l] = 3 * HW[l];
    Kk[l] = N3[l] < 1000 ? N3[l] : 1000;
    koff[l] = ko; ko += Kk[l];
  }
  // ko == KTOT == 3507

  // --- workspace carve ---
  size_t cur = 0;
  auto alloc = [&](size_t bytes) -> void* {
    cur = (cur + 255) & ~(size_t)255;
    void* p = (char*)d_ws + cur;
    cur += bytes;
    return p;
  };
  _Float16* wpack16 = (_Float16*)alloc((size_t)72 * 16 * 32 * 16 * sizeof(_Float16));
  _Float16* fcw16   = (_Float16*)alloc((size_t)1408 * 12544 * sizeof(_Float16));
  _Float16* featp[4];
  for (int l = 0; l < 4; ++l)
    featp[l] = (_Float16*)alloc((size_t)B * (Hs[l] + 2) * (Hs[l] + 2) * 256 * sizeof(_Float16));
  _Float16* t16[4];
  for (int l = 0; l < 4; ++l)
    t16[l] = (_Float16*)alloc((size_t)B * 256 * HW[l] * sizeof(_Float16));
  long loffs[4], doffs[4];
  long ltot = 0, dtot = 0;
  for (int l = 0; l < 4; ++l) { loffs[l] = ltot; ltot += (long)B * N3[l]; }
  for (int l = 0; l < 4; ++l) { doffs[l] = dtot; dtot += (long)B * N3[l] * 4; }
  float* logits_all = (float*)alloc((size_t)ltot * sizeof(float));
  float* deltas_all = (float*)alloc((size_t)dtot * sizeof(float));
  float* tscore   = (float*)alloc((size_t)B * KTOT * sizeof(float));
  int*   tidx     = (int*)  alloc((size_t)B * KTOT * sizeof(int));
  float* cscore   = (float*)alloc((size_t)B * KTOT * sizeof(float));
  float* cbox     = (float*)alloc((size_t)B * KTOT * 4 * sizeof(float));
  float* cbox_off = (float*)alloc((size_t)B * KTOT * 4 * sizeof(float));
  int*   oidx     = (int*)  alloc((size_t)B * POST * sizeof(int));
  int*   oval     = (int*)  alloc((size_t)B * POST * sizeof(int));
  float* rois     = (float*)alloc((size_t)B * POST * 4 * sizeof(float));
  _Float16* rf16  = (_Float16*)alloc((size_t)640 * 12544 * sizeof(_Float16)); // padded M

  // --- 1) weight / input conversion ---
  {
    int n = 72 * 16 * 32 * 16;
    cvt_conv_w_kernel<<<(n + 255) / 256, 256, 0, stream>>>(conv_w, wpack16, n);
    long nf = (long)1408 * 12544;
    cvt_fc_w_kernel<<<(unsigned)((nf + 255) / 256), 256, 0, stream>>>(fc_w, fcw16, nf);
  }

  // --- 2) per level: pad+NHWC f16, conv3x3 (WMMA), 1x1 heads ---
  for (int l = 0; l < 4; ++l) {
    long tp = (long)B * (Hs[l] + 2) * (Hs[l] + 2) * 256;
    cvt_featpad_kernel<<<(unsigned)((tp + 255) / 256), 256, 0, stream>>>(
        feats[l], featp[l], Hs[l], Hs[l], tp);
    dim3 g((HW[l] + 63) / 64, 4, B);
    conv3x3_wmma_kernel<<<g, 256, 0, stream>>>(featp[l], (const v16h*)wpack16, conv_b,
                                               t16[l], Hs[l], Hs[l], HW[l]);
    int total = B * 15 * HW[l];
    head1x1_kernel<<<(total + 255) / 256, 256, 0, stream>>>(
        t16[l], cls_w, cls_b, bbox_w, bbox_b,
        logits_all + loffs[l], deltas_all + doffs[l], HW[l], total);
  }

  // --- 3) top-k per (lvl, b) ---
  {
    TkParams tp;
    for (int l = 0; l < 4; ++l) { tp.loff[l] = loffs[l]; tp.N[l] = N3[l]; tp.K[l] = Kk[l]; tp.koff[l] = koff[l]; }
    topk_kernel<<<dim3(4, B), 1024, 0, stream>>>(logits_all, tp, tscore, tidx);
  }

  // --- 4) decode + clip + filter ---
  {
    DecParams dp;
    for (int l = 0; l < 4; ++l) { dp.doff[l] = doffs[l]; dp.H[l] = Hs[l]; dp.W[l] = Hs[l]; dp.size[l] = SIZES[l]; }
    int total = B * KTOT;
    decode_kernel<<<(total + 255) / 256, 256, 0, stream>>>(
        tscore, tidx, deltas_all, dp, img_h, img_w, cscore, cbox, cbox_off, total);
  }

  // --- 5) NMS, gather rois ---
  nms_kernel<<<B, 1024, 0, stream>>>(cscore, cbox_off, oidx, oval);
  gather_rois_kernel<<<(B * POST + 255) / 256, 256, 0, stream>>>(cbox, oidx, oval, rois);

  // --- 6) ROI align on feat3 (+ zero pad rows of rf) ---
  {
    long start = (long)600 * 12544, n = (long)40 * 12544;
    zero_rfpad_kernel<<<(unsigned)((n + 255) / 256), 256, 0, stream>>>(rf16, start, n);
    roi_align_kernel<<<B * POST, 256, 0, stream>>>(feats[3], rois, img_h, rf16, Hs[3], 256);
  }

  // --- 7) final FC (WMMA) ---
  {
    int M = B * POST, N = 1408, K = 12544;
    dim3 g((M + 63) / 64, N / 64);
    fc_wmma_kernel<<<g, 256, 0, stream>>>(rf16, fcw16, fc_b, (float*)d_out, M, N, K);
  }
}
